// customGRUCell_11931419149147
// MI455X (gfx1250) — matrix operand order repaired and checked by reading, not compiled, not run on hardware
//
#include <hip/hip_runtime.h>

#define H 1024
#define IN_ 512
#define B 256
#define T 256
#define DT 0.1f
#define DELTA_T 1.0f
#define Z_MIN 0.001f
#define Z_MAX 0.1f

typedef __attribute__((ext_vector_type(16))) __bf16 v16bf;
typedef __attribute__((ext_vector_type(8)))  float  v8f;
typedef __attribute__((ext_vector_type(4)))  unsigned int u32x4;

union FragU { u32x4 q[2]; v16bf f; };

__device__ __forceinline__ float sigmoidf_(float x) { return 1.0f / (1.0f + __expf(-x)); }

// round-to-nearest-even f32 -> bf16 (bit pattern as ushort)
__device__ __forceinline__ unsigned short f2bf(float f) {
    unsigned int u = __float_as_uint(f);
    u += 0x7FFFu + ((u >> 16) & 1u);
    return (unsigned short)(u >> 16);
}

// Load a 16-element bf16 fragment as two 16-byte chunks: [p, p+8) and [p+off2, p+off2+8)
__device__ __forceinline__ v16bf load_frag(const unsigned short* p, int off2) {
    FragU u;
    u.q[0] = *(const u32x4*)(p);
    u.q[1] = *(const u32x4*)(p + off2);
    return u.f;
}

// ---- CDNA5 async global->LDS copy (VGLOBAL GLOBAL_LOAD_ASYNC_TO_LDS_B128, ASYNCcnt) ----
__device__ __forceinline__ unsigned lds_off32(const void* p) {
    // generic LDS address: low 32 bits are the LDS byte offset (aperture truncation)
    return (unsigned)(unsigned long long)p;
}
__device__ __forceinline__ void async_cp16(unsigned lds_byte_off, const void* gsrc) {
    asm volatile("global_load_async_to_lds_b128 %0, %1, off"
                 :: "v"(lds_byte_off), "v"(gsrc)
                 : "memory");
}
__device__ __forceinline__ void wait_async0() {
#if __has_builtin(__builtin_amdgcn_s_wait_asynccnt)
    __builtin_amdgcn_s_wait_asynccnt(0);
#else
    asm volatile("s_wait_asynccnt 0x0" ::: "memory");
#endif
}

// ---------------- setup kernels ----------------

__global__ void prep_const_kernel(const float* __restrict__ cx, const float* __restrict__ cu,
                                  const float* __restrict__ cU,
                                  float* __restrict__ zx, float* __restrict__ zu,
                                  float* __restrict__ uc) {
    int h = blockIdx.x * blockDim.x + threadIdx.x;
    if (h < H) {
        zx[h] = Z_MIN + (Z_MAX - Z_MIN) * sigmoidf_(cx[h]);
        zu[h] = Z_MIN + (Z_MAX - Z_MIN) * sigmoidf_(cu[h]);
        uc[h] = 0.9f * sigmoidf_(cU[h]);
    }
}

__global__ void cvt_bf16_kernel(const float* __restrict__ src, unsigned short* __restrict__ dst, int n) {
    int i = blockIdx.x * blockDim.x + threadIdx.x;
    if (i < n) dst[i] = f2bf(src[i]);
}

// t = 0 state: v0 = 0 (=> r0 = 0.5), X0 = 1, U0 = 0.9; perform the first X/U update
// and emit rT/sT for the first GEMM step.
__global__ void init_state_kernel(float* __restrict__ X, float* __restrict__ U, float* __restrict__ v,
                                  const float* __restrict__ zx, const float* __restrict__ zu,
                                  const float* __restrict__ uc,
                                  unsigned short* __restrict__ rT, unsigned short* __restrict__ sT) {
    int idx = blockIdx.x * blockDim.x + threadIdx.x;
    int m = idx >> 8;        // / B
    int n = idx & (B - 1);   // % B
    float r = 0.5f;
    float zxm = zx[m], zum = zu[m], u = uc[m];
    float Xn = zxm + (1.0f - zxm) * 1.0f - DELTA_T * 0.9f * 1.0f * r;
    float Un = u * zum + (1.0f - zum) * 0.9f + DELTA_T * u * (1.0f - 0.9f) * r;
    Un = fminf(fmaxf(Un, u), 1.0f);
    X[idx] = Xn;
    U[idx] = Un;
    v[idx] = 0.0f;
    rT[n * H + m] = f2bf(r);
    sT[n * H + m] = f2bf(Un * Xn * r);
}

// per-step x_t (IN,B) -> xT (B,IN) bf16 (state independent)
__global__ void cvt_x_kernel(const float* __restrict__ x_t, unsigned short* __restrict__ xT) {
    int j = blockIdx.x * blockDim.x + threadIdx.x;
    int k = j >> 8;          // / B
    int n = j & (B - 1);     // % B
    xT[n * IN_ + k] = f2bf(x_t[(size_t)k * B + n]);
}

// ---------------- per-step fused dual-GEMM + full state-update epilogue ----------------
// 4 waves per block; block tile 64(M) x 32(N). Activation tiles (r,s then x) are staged
// into LDS via async global->LDS copies (double buffered, ASYNCcnt) and shared by all waves.
__global__ __launch_bounds__(128) void gemm_step_kernel(
        const unsigned short* __restrict__ Kbf, const unsigned short* __restrict__ Wbf,
        const unsigned short* __restrict__ Pz,  const unsigned short* __restrict__ Pr,
        const unsigned short* __restrict__ rT,  const unsigned short* __restrict__ sT,
        const unsigned short* __restrict__ xT,
        const float* __restrict__ gz, const float* __restrict__ br,
        float* __restrict__ v, float* __restrict__ out_t,
        float* __restrict__ X, float* __restrict__ U,
        const float* __restrict__ zx, const float* __restrict__ zu, const float* __restrict__ uc,
        unsigned short* __restrict__ rTn, unsigned short* __restrict__ sTn) {

    // LDS tiles: [buf][r|s][32 rows x 32 k], rows padded to 40 ushorts (80B)
    __shared__ unsigned short tile[2][2][32 * 40];

    const int tid  = threadIdx.x;
    const int wave = tid >> 5;
    const int lane = tid & 31;
    const int lsub = lane & 15;
    const int hi   = lane >> 4;
    const int wm   = wave & 1;   // M half of block tile
    const int wn   = wave >> 1;  // N tile of block tile

    const int mBlk = blockIdx.x * 64;
    const int nBlk = blockIdx.y * 32;
    const int m0   = mBlk + wm * 32;   // wave owns rows [m0, m0+32): two 16-row tiles
    const int n0   = nBlk + wn * 16;

    // staging assignment: 128 threads x 16B cover one 32x32 bf16 tile (2KB)
    const int sn = tid >> 2;            // 0..31 : row within tile
    const int sc = (tid & 3) * 8;       // 0,8,16,24 : k-element offset

    v8f accZ0 = {}, accZ1 = {}, accR0 = {}, accR1 = {};

    const unsigned short* kBase0 = Kbf + (size_t)(m0 + lsub)      * H + hi * 8;
    const unsigned short* kBase1 = Kbf + (size_t)(m0 + 16 + lsub) * H + hi * 8;
    const unsigned short* wBase0 = Wbf + (size_t)(m0 + lsub)      * H + hi * 8;
    const unsigned short* wBase1 = Wbf + (size_t)(m0 + 16 + lsub) * H + hi * 8;

    const unsigned short* rRow = rT + (size_t)(nBlk + sn) * H   + sc;
    const unsigned short* sRow = sT + (size_t)(nBlk + sn) * H   + sc;
    const unsigned short* xRow = xT + (size_t)(nBlk + sn) * IN_ + sc;

    const int ldsRow   = sn * 40 + sc;                    // staging slot (elements)
    const int fragBase = (wn * 16 + lsub) * 40 + hi * 8;  // B-frag base (elements)

    // ---- phase 1: K = H over weight matrices K and w_r; r/s tiles via LDS ----
    async_cp16(lds_off32(&tile[0][0][ldsRow]), rRow);
    async_cp16(lds_off32(&tile[0][1][ldsRow]), sRow);
    const int NK1 = H / 32;
    for (int i = 0; i < NK1; ++i) {
        const int kk = i * 32, cur = i & 1;
        wait_async0();
        __syncthreads();
        if (i + 1 < NK1) {
            async_cp16(lds_off32(&tile[cur ^ 1][0][ldsRow]), rRow + kk + 32);
            async_cp16(lds_off32(&tile[cur ^ 1][1][ldsRow]), sRow + kk + 32);
        }
        v16bf bR  = load_frag(&tile[cur][0][fragBase], 16);
        v16bf bS  = load_frag(&tile[cur][1][fragBase], 16);
        v16bf aK0 = load_frag(kBase0 + kk, 16);
        v16bf aK1 = load_frag(kBase1 + kk, 16);
        v16bf aW0 = load_frag(wBase0 + kk, 16);
        v16bf aW1 = load_frag(wBase1 + kk, 16);
        accZ0 = __builtin_amdgcn_wmma_f32_16x16x32_bf16(false, aK0, false, bR, (short)0, accZ0, false, false);
        accZ1 = __builtin_amdgcn_wmma_f32_16x16x32_bf16(false, aK1, false, bR, (short)0, accZ1, false, false);
        accR0 = __builtin_amdgcn_wmma_f32_16x16x32_bf16(false, aW0, false, bS, (short)0, accR0, false, false);
        accR1 = __builtin_amdgcn_wmma_f32_16x16x32_bf16(false, aW1, false, bS, (short)0, accR1, false, false);
    }
    __syncthreads();

    // ---- phase 2: K = IN over p_z and p_r; shared x tile via LDS ----
    const unsigned short* pzBase0 = Pz + (size_t)(m0 + lsub)      * IN_ + hi * 8;
    const unsigned short* pzBase1 = Pz + (size_t)(m0 + 16 + lsub) * IN_ + hi * 8;
    const unsigned short* prBase0 = Pr + (size_t)(m0 + lsub)      * IN_ + hi * 8;
    const unsigned short* prBase1 = Pr + (size_t)(m0 + 16 + lsub) * IN_ + hi * 8;

    async_cp16(lds_off32(&tile[0][0][ldsRow]), xRow);
    const int NK2 = IN_ / 32;
    for (int i = 0; i < NK2; ++i) {
        const int kk = i * 32, cur = i & 1;
        wait_async0();
        __syncthreads();
        if (i + 1 < NK2) {
            async_cp16(lds_off32(&tile[cur ^ 1][0][ldsRow]), xRow + kk + 32);
        }
        v16bf bX   = load_frag(&tile[cur][0][fragBase], 16);
        v16bf aPz0 = load_frag(pzBase0 + kk, 16);
        v16bf aPz1 = load_frag(pzBase1 + kk, 16);
        v16bf aPr0 = load_frag(prBase0 + kk, 16);
        v16bf aPr1 = load_frag(prBase1 + kk, 16);
        accZ0 = __builtin_amdgcn_wmma_f32_16x16x32_bf16(false, aPz0, false, bX, (short)0, accZ0, false, false);
        accZ1 = __builtin_amdgcn_wmma_f32_16x16x32_bf16(false, aPz1, false, bX, (short)0, accZ1, false, false);
        accR0 = __builtin_amdgcn_wmma_f32_16x16x32_bf16(false, aPr0, false, bX, (short)0, accR0, false, false);
        accR1 = __builtin_amdgcn_wmma_f32_16x16x32_bf16(false, aPr1, false, bX, (short)0, accR1, false, false);
    }

    // ---- fused epilogue: v update, output store, and next-step r/s/X/U ----
    const int n = n0 + lsub;
#pragma unroll
    for (int i = 0; i < 8; ++i) {
        {
            const int m = m0 + hi * 8 + i;
            const size_t vi = (size_t)m * B + n;
            float zt = DT * sigmoidf_(accZ0[i] + gz[m]);
            float vn = (1.0f - zt) * v[vi] + DT * (accR0[i] + br[m]);
            v[vi] = vn;
            out_t[(size_t)n * H + m] = vn;
            float r  = sigmoidf_(vn);
            float Xo = X[vi], Uo = U[vi];
            float zxm = zx[m], zum = zu[m], u = uc[m];
            float Xn = zxm + (1.0f - zxm) * Xo - DELTA_T * Uo * Xo * r;
            float Un = u * zum + (1.0f - zum) * Uo + DELTA_T * u * (1.0f - Uo) * r;
            Un = fminf(fmaxf(Un, u), 1.0f);
            X[vi] = Xn; U[vi] = Un;
            rTn[n * H + m] = f2bf(r);
            sTn[n * H + m] = f2bf(Un * Xn * r);
        }
        {
            const int m = m0 + 16 + hi * 8 + i;
            const size_t vi = (size_t)m * B + n;
            float zt = DT * sigmoidf_(accZ1[i] + gz[m]);
            float vn = (1.0f - zt) * v[vi] + DT * (accR1[i] + br[m]);
            v[vi] = vn;
            out_t[(size_t)n * H + m] = vn;
            float r  = sigmoidf_(vn);
            float Xo = X[vi], Uo = U[vi];
            float zxm = zx[m], zum = zu[m], u = uc[m];
            float Xn = zxm + (1.0f - zxm) * Xo - DELTA_T * Uo * Xo * r;
            float Un = u * zum + (1.0f - zum) * Uo + DELTA_T * u * (1.0f - Uo) * r;
            Un = fminf(fmaxf(Un, u), 1.0f);
            X[vi] = Xn; U[vi] = Un;
            rTn[n * H + m] = f2bf(r);
            sTn[n * H + m] = f2bf(Un * Xn * r);
        }
    }
}

// ---------------- host launcher ----------------

extern "C" void kernel_launch(void* const* d_in, const int* in_sizes, int n_in,
                              void* d_out, int out_size, void* d_ws, size_t ws_size,
                              hipStream_t stream) {
    const float* x   = (const float*)d_in[0];
    const float* c_x = (const float*)d_in[1];
    const float* c_u = (const float*)d_in[2];
    const float* c_U = (const float*)d_in[3];
    const float* w_r = (const float*)d_in[4];
    const float* p_r = (const float*)d_in[5];
    const float* b_r = (const float*)d_in[6];
    const float* g_z = (const float*)d_in[7];
    const float* Kw  = (const float*)d_in[8];
    const float* p_z = (const float*)d_in[9];
    float* out = (float*)d_out;

    char* ws = (char*)d_ws;
    size_t off = 0;
    auto carve = [&](size_t bytes) { void* p = ws + off; off += (bytes + 255) & ~(size_t)255; return p; };
    unsigned short* Kbf = (unsigned short*)carve((size_t)H * H * 2);
    unsigned short* Wbf = (unsigned short*)carve((size_t)H * H * 2);
    unsigned short* Pz  = (unsigned short*)carve((size_t)H * IN_ * 2);
    unsigned short* Pr  = (unsigned short*)carve((size_t)H * IN_ * 2);
    unsigned short* rT  = (unsigned short*)carve((size_t)B * H * 2);
    unsigned short* sT  = (unsigned short*)carve((size_t)B * H * 2);
    unsigned short* xT  = (unsigned short*)carve((size_t)B * IN_ * 2);
    float* X  = (float*)carve((size_t)H * B * 4);
    float* U  = (float*)carve((size_t)H * B * 4);
    float* v  = (float*)carve((size_t)H * B * 4);
    float* zx = (float*)carve((size_t)H * 4);
    float* zu = (float*)carve((size_t)H * 4);
    float* uc = (float*)carve((size_t)H * 4);

    prep_const_kernel<<<(H + 255) / 256, 256, 0, stream>>>(c_x, c_u, c_U, zx, zu, uc);
    cvt_bf16_kernel<<<(H * H   + 255) / 256, 256, 0, stream>>>(Kw,  Kbf, H * H);
    cvt_bf16_kernel<<<(H * H   + 255) / 256, 256, 0, stream>>>(w_r, Wbf, H * H);
    cvt_bf16_kernel<<<(H * IN_ + 255) / 256, 256, 0, stream>>>(p_z, Pz, H * IN_);
    cvt_bf16_kernel<<<(H * IN_ + 255) / 256, 256, 0, stream>>>(p_r, Pr, H * IN_);
    init_state_kernel<<<(H * B) / 256, 256, 0, stream>>>(X, U, v, zx, zu, uc, rT, sT);

    for (int t = 0; t < T; ++t) {
        cvt_x_kernel<<<(IN_ * B) / 256, 256, 0, stream>>>(x + (size_t)t * IN_ * B, xT);
        gemm_step_kernel<<<dim3(H / 64, B / 32), 128, 0, stream>>>(
            Kbf, Wbf, Pz, Pr, rT, sT, xT, g_z, b_r, v, out + (size_t)t * B * H,
            X, U, zx, zu, uc, rT, sT);
    }
}
